// VisionMambaV2_19061064859765
// MI455X (gfx1250) — matrix operand order repaired and checked by reading, not compiled, not run on hardware
//
#include <hip/hip_runtime.h>
#include <hip/hip_bf16.h>

#define DM  384
#define DI  768
#define NSz 16
#define DEPTHC 2
#define BB  4
#define LLq 256
#define TOK (BB*LLq)   // 1024 tokens

typedef __attribute__((ext_vector_type(16))) _Float16 v16h;
typedef __attribute__((ext_vector_type(8)))  _Float16 v8h;
typedef __attribute__((ext_vector_type(8)))  float    v8f;

#define WMMA_F32_F16(a, b, c) \
  __builtin_amdgcn_wmma_f32_16x16x32_f16(false, (a), false, (b), (short)0, (c), false, false)

// ---------------------------------------------------------------- converts
__global__ void cvt_f32_to_f16_k(const float* __restrict__ in, _Float16* __restrict__ out, int n){
  int i = blockIdx.x*blockDim.x + threadIdx.x;
  if (i < n) out[i] = (_Float16)in[i];
}

// ---------------------------------------------------------------- patch embed
// h[b,l,c] : l==0 -> cls, else 16x16 conv patch (grid 15x17), + pos_embed
__global__ void patch_embed_k(const float* __restrict__ x, const float* __restrict__ pw,
                              const float* __restrict__ pb, const float* __restrict__ cls,
                              const float* __restrict__ pos, float* __restrict__ h){
  int idx = blockIdx.x*blockDim.x + threadIdx.x;          // < 4*256*384
  if (idx >= TOK*DM) return;
  int c = idx % DM;
  int l = (idx / DM) % LLq;
  int b = idx / (DM*LLq);
  float v;
  if (l == 0) {
    v = cls[c];
  } else {
    int p = l - 1, ph = p / 17, pwd = p % 17;
    float s = 0.f;
    for (int ci = 0; ci < 3; ++ci)
      for (int kh = 0; kh < 16; ++kh) {
        const float* xr = x  + ((size_t)(b*3+ci)*240 + ph*16 + kh)*272 + pwd*16;
        const float* wr = pw + ((size_t)(c*3+ci)*16  + kh)*16;
        #pragma unroll
        for (int kw = 0; kw < 16; ++kw) s += xr[kw]*wr[kw];
      }
    v = s + pb[c];
  }
  h[idx] = v + pos[l*DM + c];
}

// ---------------------------------------------------------------- layernorm (384ch, one wave/token)
__global__ void layernorm_k(const float* __restrict__ x, const float* __restrict__ w,
                            const float* __restrict__ bvec,
                            _Float16* __restrict__ out16, float* __restrict__ out32, int ntok){
  int wave = threadIdx.x >> 5, lane = threadIdx.x & 31;
  int tok = blockIdx.x * (blockDim.x >> 5) + wave;
  if (tok >= ntok) return;
  const float* row = x + (size_t)tok*DM;
  float vals[12];
  float s = 0.f;
  #pragma unroll
  for (int j = 0; j < 12; ++j) { vals[j] = row[lane + 32*j]; s += vals[j]; }
  #pragma unroll
  for (int o = 16; o; o >>= 1) s += __shfl_xor(s, o, 32);
  float mean = s * (1.f/DM);
  float v = 0.f;
  #pragma unroll
  for (int j = 0; j < 12; ++j) { float d = vals[j]-mean; v += d*d; }
  #pragma unroll
  for (int o = 16; o; o >>= 1) v += __shfl_xor(v, o, 32);
  float rstd = rsqrtf(v*(1.f/DM) + 1e-5f);
  #pragma unroll
  for (int j = 0; j < 12; ++j) {
    int c = lane + 32*j;
    float y = (vals[j]-mean)*rstd*w[c] + bvec[c];
    if (out16) out16[(size_t)tok*DM + c] = (_Float16)y;
    if (out32) out32[(size_t)tok*DM + c] = y;
  }
}

// ---------------------------------------------------------------- WMMA NT GEMM, 16x64 tile/wave
// C[M,N] = A[M,K] * B[N,K]^T (+bias[N]) (+res[M,N]); A,B f16 (K contiguous), C f32.
// One wave computes a 16x64 tile. Software-pipelined double buffer: next-K fragments
// are loaded before the WMMAs consuming the current fragments, keeping both sets live
// so loads overlap matrix math instead of draining loadcnt before every WMMA.
// Requires: K % 32 == 0, N % 64 == 0, M % 128 == 0 (blockDim = {32,8}).
__global__ void gemm_nt_wmma_k(const _Float16* __restrict__ A, const _Float16* __restrict__ B,
                               const float* __restrict__ bias, const float* __restrict__ res,
                               float* __restrict__ C, int M, int N, int K){
  int lane = threadIdx.x;                  // 0..31
  int m_tile = blockIdx.y * blockDim.y + threadIdx.y;
  if (m_tile * 16 >= M) return;            // wave-uniform
  int n_base = blockIdx.x * 64;
  int hv = lane >> 4, lr = lane & 15;
  const _Float16* Arow  = A + (size_t)(m_tile*16 + lr) * K;
  const _Float16* Brow0 = B + (size_t)(n_base +      lr) * K;
  const _Float16* Brow1 = B + (size_t)(n_base + 16 + lr) * K;
  const _Float16* Brow2 = B + (size_t)(n_base + 32 + lr) * K;
  const _Float16* Brow3 = B + (size_t)(n_base + 48 + lr) * K;

  // A layout: lane<16 holds K{k..k+7, k+16..k+23}; lane>=16 holds K{k+8..15, k+24..31}
  auto ldA = [&](int k) -> v16h {
    v8h x0 = *(const v8h*)(Arow + k + hv*8);
    v8h x1 = *(const v8h*)(Arow + k + hv*8 + 16);
    v16h r;
    #pragma unroll
    for (int i = 0; i < 8; ++i) { r[i] = x0[i]; r[i+8] = x1[i]; }
    return r;
  };
  // B layout: lane holds column lr, K{k+8h..k+8h+7, k+16+8h..k+16+8h+7}, h = lane/16
  auto ldB = [&](const _Float16* Br, int k) -> v16h {
    v8h y0 = *(const v8h*)(Br + k + hv*8);
    v8h y1 = *(const v8h*)(Br + k + hv*8 + 16);
    v16h r;
    #pragma unroll
    for (int i = 0; i < 8; ++i) { r[i] = y0[i]; r[i+8] = y1[i]; }
    return r;
  };

  v8f acc0 = {}, acc1 = {}, acc2 = {}, acc3 = {};
  v16h a  = ldA(0);
  v16h b0 = ldB(Brow0, 0), b1 = ldB(Brow1, 0), b2 = ldB(Brow2, 0), b3 = ldB(Brow3, 0);

  int k = 0;
  for (; k + 32 < K; k += 32) {
    int kn = k + 32;
    __builtin_prefetch(Arow  + kn + 64, 0, 1);
    __builtin_prefetch(Brow0 + kn + 64, 0, 1);
    __builtin_prefetch(Brow1 + kn + 64, 0, 1);
    __builtin_prefetch(Brow2 + kn + 64, 0, 1);
    __builtin_prefetch(Brow3 + kn + 64, 0, 1);
    v16h an  = ldA(kn);
    v16h b0n = ldB(Brow0, kn), b1n = ldB(Brow1, kn), b2n = ldB(Brow2, kn), b3n = ldB(Brow3, kn);
    acc0 = WMMA_F32_F16(a, b0, acc0);
    acc1 = WMMA_F32_F16(a, b1, acc1);
    acc2 = WMMA_F32_F16(a, b2, acc2);
    acc3 = WMMA_F32_F16(a, b3, acc3);
    a = an; b0 = b0n; b1 = b1n; b2 = b2n; b3 = b3n;
  }
  acc0 = WMMA_F32_F16(a, b0, acc0);
  acc1 = WMMA_F32_F16(a, b1, acc1);
  acc2 = WMMA_F32_F16(a, b2, acc2);
  acc3 = WMMA_F32_F16(a, b3, acc3);

  #pragma unroll
  for (int j = 0; j < 4; ++j) {
    v8f acc = (j == 0) ? acc0 : (j == 1) ? acc1 : (j == 2) ? acc2 : acc3;
    int col = n_base + j*16 + lr;
    float bv = bias ? bias[col] : 0.f;
    #pragma unroll
    for (int r = 0; r < 8; ++r) {
      int row = m_tile*16 + hv*8 + r;
      size_t o = (size_t)row*N + col;
      float v = acc[r] + bv;
      if (res) v += res[o];
      C[o] = v;
    }
  }
}

// ---------------------------------------------------------------- depthwise 3x3 + bias + SiLU
__global__ void dwconv_silu_k(const float* __restrict__ xz, const float* __restrict__ cw,
                              const float* __restrict__ cb, float* __restrict__ u32,
                              _Float16* __restrict__ u16){
  int idx = blockIdx.x*blockDim.x + threadIdx.x;           // < 4*256*768
  if (idx >= TOK*DI) return;
  int c = idx % DI;
  int l = (idx / DI) % LLq;
  int b = idx / (DI*LLq);
  int hh = l >> 4, ww = l & 15;
  float s = 0.f;
  #pragma unroll
  for (int kh = 0; kh < 3; ++kh) {
    int y = hh + kh - 1; if ((unsigned)y >= 16u) continue;
    #pragma unroll
    for (int kw = 0; kw < 3; ++kw) {
      int xw2 = ww + kw - 1; if ((unsigned)xw2 >= 16u) continue;
      s += xz[((size_t)(b*LLq + (y<<4) + xw2))*(2*DI) + c] * cw[c*9 + kh*3 + kw];
    }
  }
  s += cb[c];
  float r = s / (1.f + expf(-s));
  u32[idx] = r;
  u16[idx] = (_Float16)r;
}

// ---------------------------------------------------------------- dt mean over 4 groups
__global__ void dt_mean_k(const float* __restrict__ xdbl, _Float16* __restrict__ dt16){
  int idx = blockIdx.x*blockDim.x + threadIdx.x;           // < 1024*384
  if (idx >= TOK*DM) return;
  int c = idx % DM;
  int row = idx / DM;
  const float* r = xdbl + (size_t)row*(4*DM + 8*NSz);
  float s = r[c] + r[DM + c] + r[2*DM + c] + r[3*DM + c];
  dt16[idx] = (_Float16)(s*0.25f);
}

// ---------------------------------------------------------------- direction map (seq l -> token)
__device__ __forceinline__ int tok_map(int dir, int l){
  int a = l >> 4, c = l & 15;
  switch (dir) {
    case 0:  return l;                    // row-major
    case 1:  return (a<<4) | (15 - c);    // reversed w
    case 2:  return (c<<4) | a;           // transposed
    default: return ((15 - c)<<4) | a;    // reversed h, transposed
  }
}

// ---------------------------------------------------------------- selective scan
// One 16-lane group per (dir,b,ch); wave covers 2 channels. Replicates the reference's
// dA_cs = exp(suffix-sum of exp(delta*A)) via T - prefix form; y accum with atomics.
// Prefetches 8 sequence steps ahead to cover the serial-dependence load latency.
__global__ void scan_k(const float* __restrict__ delta,  // (1024, 3072)
                       const float* __restrict__ u,      // (1024, 768)
                       const float* __restrict__ xdbl,   // (1024, 1664)
                       const float* __restrict__ A_log,  // (768,16)
                       const float* __restrict__ Dp,     // (768)
                       float* __restrict__ yacc){        // (1024, 768)
  int wave = (blockIdx.x*blockDim.x + threadIdx.x) >> 5;  // < 6144
  int lane = threadIdx.x & 31;
  int dir = wave / (BB*384);
  int rem = wave % (BB*384);
  int b   = rem / 384;
  int ch  = (rem % 384)*2 + (lane >> 4);
  int n   = lane & 15;
  const int XW = 4*DM + 8*NSz;            // 1664
  float Ac = -expf(A_log[ch*NSz + n]);
  float Dc = Dp[ch];
  // pass 1: total sum of dA over the sequence
  float T = 0.f;
  for (int l = 0; l < LLq; ++l) {
    int row = b*LLq + tok_map(dir, l);
    if (l + 8 < LLq) {
      int rowp = b*LLq + tok_map(dir, l + 8);
      __builtin_prefetch(&delta[(size_t)rowp*(4*DI) + dir*DI + ch], 0, 1);
    }
    float del = delta[(size_t)row*(4*DI) + dir*DI + ch];
    T += expf(del*Ac);
  }
  // pass 2: suffix S = T - prefix; xs = cumsum(dB_u*exp(S))/(exp(S)+eps); y = <xs,C> + u*D
  float R = 0.f, P = 0.f;
  for (int l = 0; l < LLq; ++l) {
    int row = b*LLq + tok_map(dir, l);
    if (l + 8 < LLq) {
      int rowp = b*LLq + tok_map(dir, l + 8);
      __builtin_prefetch(&u[(size_t)rowp*DI + ch], 0, 1);
      __builtin_prefetch(&xdbl[(size_t)rowp*XW + 4*DM + dir*NSz + n], 0, 1);
    }
    float del = delta[(size_t)row*(4*DI) + dir*DI + ch];
    float uu  = u[(size_t)row*DI + ch];
    float dA  = expf(del*Ac);
    R += dA;
    float dacs = expf(T - R);
    float bn = xdbl[(size_t)row*XW + 4*DM + dir*NSz + n];
    float cn = xdbl[(size_t)row*XW + 4*DM + 4*NSz + dir*NSz + n];
    P += del*uu*bn*dacs;
    float contrib = (P / (dacs + 1e-12f)) * cn;
    contrib += __shfl_xor(contrib, 8, 32);
    contrib += __shfl_xor(contrib, 4, 32);
    contrib += __shfl_xor(contrib, 2, 32);
    contrib += __shfl_xor(contrib, 1, 32);
    if (n == 0) atomicAdd(&yacc[(size_t)row*DI + ch], contrib + uu*Dc);
  }
}

// ---------------------------------------------------------------- gate: y/4 * silu(z) -> f16
__global__ void gate_k(const float* __restrict__ yacc, const float* __restrict__ xz,
                       _Float16* __restrict__ yg){
  int idx = blockIdx.x*blockDim.x + threadIdx.x;           // < 1024*768
  if (idx >= TOK*DI) return;
  int c = idx % DI;
  int row = idx / DI;
  float z = xz[(size_t)row*(2*DI) + DI + c];
  float sz = z / (1.f + expf(-z));
  yg[idx] = (_Float16)(yacc[idx]*0.25f*sz);
}

// ================================================================ host
extern "C" void kernel_launch(void* const* d_in, const int* in_sizes, int n_in,
                              void* d_out, int out_size, void* d_ws, size_t ws_size,
                              hipStream_t stream) {
  (void)in_sizes; (void)n_in; (void)out_size; (void)ws_size;
  const float* x_in  = (const float*)d_in[0];
  const float* pw    = (const float*)d_in[1];
  const float* pb    = (const float*)d_in[2];
  const float* cls   = (const float*)d_in[3];
  const float* pos   = (const float*)d_in[4];
  const float* nw    = (const float*)d_in[5];
  const float* nb    = (const float*)d_in[6];
  const float* inw   = (const float*)d_in[7];
  const float* cw    = (const float*)d_in[8];
  const float* cb    = (const float*)d_in[9];
  const float* xw    = (const float*)d_in[10];
  const float* dtw   = (const float*)d_in[11];
  const float* dtb   = (const float*)d_in[12];
  const float* alog  = (const float*)d_in[13];
  const float* dpar  = (const float*)d_in[14];
  const float* ow    = (const float*)d_in[15];
  const float* fnw   = (const float*)d_in[16];
  const float* fnb   = (const float*)d_in[17];

  char* ws = (char*)d_ws;
  size_t off = 0;
  auto alloc = [&](size_t bytes) { size_t o = off; off = (off + bytes + 255) & ~(size_t)255; return o; };

  float*    h     = (float*)   (ws + alloc((size_t)TOK*DM*4));
  _Float16* xn16  = (_Float16*)(ws + alloc((size_t)TOK*DM*2));
  float*    xz    = (float*)   (ws + alloc((size_t)TOK*2*DI*4));
  float*    u32   = (float*)   (ws + alloc((size_t)TOK*DI*4));
  _Float16* u16   = (_Float16*)(ws + alloc((size_t)TOK*DI*2));
  float*    xdbl  = (float*)   (ws + alloc((size_t)TOK*1664*4));
  _Float16* dt16  = (_Float16*)(ws + alloc((size_t)TOK*DM*2));
  float*    delta = (float*)   (ws + alloc((size_t)TOK*4*DI*4));
  float*    yacc  = (float*)   (ws + alloc((size_t)TOK*DI*4));
  _Float16* yg16  = (_Float16*)(ws + alloc((size_t)TOK*DI*2));
  _Float16* wIn16 = (_Float16*)(ws + alloc((size_t)(2*DI)*DM*2));
  _Float16* wX16  = (_Float16*)(ws + alloc((size_t)1664*DI*2));
  _Float16* wDt16 = (_Float16*)(ws + alloc((size_t)(4*DI)*DM*2));
  _Float16* wOw16 = (_Float16*)(ws + alloc((size_t)DM*DI*2));

  auto cvt = [&](const float* src, _Float16* dst, int n) {
    cvt_f32_to_f16_k<<<(n + 255)/256, 256, 0, stream>>>(src, dst, n);
  };
  auto gemm = [&](const _Float16* A, const _Float16* B, const float* bias,
                  const float* res, float* C, int M, int N, int K) {
    dim3 blk(32, 8), grd(N/64, M/128);
    gemm_nt_wmma_k<<<grd, blk, 0, stream>>>(A, B, bias, res, C, M, N, K);
  };

  // patch embed + cls + pos -> h (1024 x 384)
  patch_embed_k<<<(TOK*DM + 255)/256, 256, 0, stream>>>(x_in, pw, pb, cls, pos, h);

  for (int i = 0; i < DEPTHC; ++i) {
    cvt(inw + (size_t)i*(2*DI)*DM,  wIn16, 2*DI*DM);
    cvt(xw  + (size_t)i*1664*DI,    wX16,  1664*DI);
    cvt(dtw + (size_t)i*(4*DI)*DM,  wDt16, 4*DI*DM);
    cvt(ow  + (size_t)i*DM*DI,      wOw16, DM*DI);

    layernorm_k<<<TOK/8, 256, 0, stream>>>(h, nw + i*DM, nb + i*DM, xn16, (float*)nullptr, TOK);
    gemm(xn16, wIn16, nullptr, nullptr, xz, TOK, 2*DI, DM);                 // in_proj
    dwconv_silu_k<<<(TOK*DI + 255)/256, 256, 0, stream>>>(xz, cw + (size_t)i*DI*9, cb + i*DI, u32, u16);
    gemm(u16, wX16, nullptr, nullptr, xdbl, TOK, 1664, DI);                 // x_proj
    dt_mean_k<<<(TOK*DM + 255)/256, 256, 0, stream>>>(xdbl, dt16);
    gemm(dt16, wDt16, dtb + i*4*DI, nullptr, delta, TOK, 4*DI, DM);         // dt_proj (+bias)
    hipMemsetAsync(yacc, 0, (size_t)TOK*DI*4, stream);
    scan_k<<<(6144*32)/256, 256, 0, stream>>>(delta, u32, xdbl,
                                              alog + (size_t)i*DI*NSz, dpar + i*DI, yacc);
    gate_k<<<(TOK*DI + 255)/256, 256, 0, stream>>>(yacc, xz, yg16);
    gemm(yg16, wOw16, nullptr, h, h, TOK, DM, DI);                          // out_proj + residual
  }

  layernorm_k<<<TOK/8, 256, 0, stream>>>(h, fnw, fnb, (_Float16*)nullptr, (float*)d_out, TOK);
}
